// CrossAttention_8495445311783
// MI455X (gfx1250) — matrix operand order repaired and checked by reading, not compiled, not run on hardware
//
#include <hip/hip_runtime.h>

typedef _Float16 half_t;
typedef __attribute__((ext_vector_type(16))) _Float16 v16h;
typedef __attribute__((ext_vector_type(8)))  float    v8f;

#define DIMC  64
#define CTXC  128
#define NB    4
#define DD    64
#define HH    64
#define WW    64
#define DHW   (DD * HH * WW)   /* 262144 */
#define HW    (HH * WW)        /* 4096  */
#define VLD   72               /* padded ld for transposed V tile */
#define XPAD  8                /* padding for transposed X tile   */

// ---------------------------------------------------------------------------
// CDNA5 wave32 WMMA fragment layouts (CDNA5 ISA):
//  A 16x32 f16 : lane l -> M = l%16 ; element e -> K = 16*(l/16) + e (k order used here, not ISA table)
//  B 32x16 f16 : lane l -> N = l%16 ; element e -> K = 16*(l/16) + e (k order used here, not ISA table)
//  C/D 16x16 f32: vgpr r, lanes 0-15: M=r, N=lane ; lanes 16-31: M=r+8, N=lane-16
// All LDS tiles are laid out so fragment elements are CONTIGUOUS in e
// (vectorizes to ds_load_b128 instead of 16x ds_load_u16).
// ---------------------------------------------------------------------------
__device__ __forceinline__ v16h load_a_frag(const half_t* s, int ld, int row0,
                                            int col0, int lane) {
  v16h a;
  const int m  = lane & 15;
  const int kb = (lane >> 4) << 4;          // 0 or 16
#pragma unroll
  for (int e = 0; e < 16; ++e) {
    const int k = kb + e;
    a[e] = s[(row0 + m) * ld + col0 + k];
  }
  return a;
}

// B fragment where the source is stored K-contiguous: s[n][k] = B[k][n]
__device__ __forceinline__ v16h load_bt_frag(const half_t* s, int ld, int krow0,
                                             int col0, int lane) {
  v16h b;
  const int n  = lane & 15;
  const int kb = (lane >> 4) << 4;          // 0 or 16
#pragma unroll
  for (int e = 0; e < 16; ++e)
    b[e] = s[(col0 + n) * ld + krow0 + kb + e];
  return b;
}

__device__ __forceinline__ v8f wmma16(v16h a, v16h b, v8f c) {
  return __builtin_amdgcn_wmma_f32_16x16x32_f16(false, a, false, b, (short)0, c,
                                                false, false);
}

// ---------------------------------------------------------------------------
// Kernel 1: 1x1x1 conv projection, f16 WMMA GEMM [64 x Cin] x [Cin x 262144].
// oscale folds the attention softmax scale into the q projection.
// grid = (DHW/128, B), block = 256 (8 waves). LDS ~ 50 KB.
// ---------------------------------------------------------------------------
__global__ __launch_bounds__(256)
void proj16_kernel(const float* __restrict__ src, const float* __restrict__ w,
                   const float* __restrict__ bias, half_t* __restrict__ dst,
                   int Cin, float oscale) {
  __shared__ half_t sW[DIMC * CTXC];           // [m][k] row-major, ld = Cin
  __shared__ half_t sXT[128 * (CTXC + XPAD)];  // [n][k] transposed, ld = Cin+XPAD

  const int tid = threadIdx.x;
  const int b   = blockIdx.y;
  const int n0  = blockIdx.x * 128;
  const int ldx = Cin + XPAD;
  const float* srcb = src + (size_t)b * Cin * DHW;

  for (int i = tid; i < DIMC * Cin; i += 256) sW[i] = (half_t)w[i];
  for (int i = tid; i < Cin * 128; i += 256) {
    const int c = i >> 7, n = i & 127;
    sXT[n * ldx + c] = (half_t)srcb[(size_t)c * DHW + n0 + n];
  }
  if (n0 + 128 < DHW)  // gfx1250 global_prefetch_b8 for the next N-tile
    __builtin_prefetch(srcb + (size_t)(tid & (Cin - 1)) * DHW + n0 + 128, 0, 0);
  __syncthreads();

  const int wave = tid >> 5, lane = tid & 31;
  const int mt  = wave >> 1;          // 4 M-tiles, 2 waves each
  const int ntb = (wave & 1) * 4;     // 4 N-tiles per wave
  half_t* db = dst + (size_t)b * DIMC * DHW;

  const v8f vzero = {};
  v8f acc[4];
#pragma unroll
  for (int j = 0; j < 4; ++j) acc[j] = vzero;

  for (int kk = 0; kk < Cin; kk += 32) {
    v16h a = load_a_frag(sW, Cin, mt * 16, kk, lane);   // shared across N-tiles
#pragma unroll
    for (int j = 0; j < 4; ++j) {
      v16h bf = load_bt_frag(sXT, ldx, kk, (ntb + j) * 16, lane);
      acc[j] = wmma16(a, bf, acc[j]);
    }
  }

  const int mh = (lane >> 4) << 3;
#pragma unroll
  for (int j = 0; j < 4; ++j) {
    const int n = (ntb + j) * 16 + (lane & 15);
#pragma unroll
    for (int r = 0; r < 8; ++r) {
      const int m = mt * 16 + r + mh;
      db[(size_t)m * DHW + n0 + n] = (half_t)((acc[j][r] + bias[m]) * oscale);
    }
  }
}

// ---------------------------------------------------------------------------
// Kernel 2: per-(b,c,d) 64x64 attention + residual. One wave per channel,
// 2 waves per block. Q/K staged with GLOBAL_LOAD_ASYNC_TO_LDS_B128 (ASYNCcnt),
// V staged transposed so PV B-fragments are contiguous. LDS = 50 KB.
// grid = (DIM/2, D, B), block = 64.
// ---------------------------------------------------------------------------
__global__ __launch_bounds__(64)
void attn64_kernel(const half_t* __restrict__ qws, const half_t* __restrict__ kws,
                   const half_t* __restrict__ vws, const float* __restrict__ x,
                   float* __restrict__ out) {
  __shared__ half_t sQ[2][HW];        // 8 KB per wave; reused to stage softmax(P)
  __shared__ half_t sK[2][HW];        // 8 KB per wave, row-major [g][w]
  __shared__ half_t sVT[2][WW * VLD]; // 9 KB per wave, transposed [w][g]

  const int wave = threadIdx.x >> 5, lane = threadIdx.x & 31;
  const int c = blockIdx.x * 2 + wave;
  const int d = blockIdx.y, b = blockIdx.z;
  const size_t slice = (((size_t)b * DIMC + c) * DD + d) * (size_t)HW;

  const float* xs = x + slice;
  float* os = out + slice;
  __builtin_prefetch(xs + lane * 128, 0, 0);   // warm residual tile

  // ---- async-copy Q and K straight into LDS (no VGPR round trip) ----
  {
    const uint4* gq = (const uint4*)(qws + slice);
    const uint4* gk = (const uint4*)(kws + slice);
    const unsigned qlds = (unsigned)(uintptr_t)&sQ[wave][0];
    const unsigned klds = (unsigned)(uintptr_t)&sK[wave][0];
    for (int i = lane; i < HW / 8; i += 32) {
      asm volatile("global_load_async_to_lds_b128 %0, %1, off"
                   :: "v"(qlds + (unsigned)i * 16u), "v"(gq + i) : "memory");
      asm volatile("global_load_async_to_lds_b128 %0, %1, off"
                   :: "v"(klds + (unsigned)i * 16u), "v"(gk + i) : "memory");
    }
  }
  // ---- V staged transposed via VGPRs (transpose needs element stores) ----
  {
    const uint4* gv = (const uint4*)(vws + slice);
    for (int j = lane; j < HW / 8; j += 32) {
      const int g  = j >> 3;
      const int w0 = (j & 7) * 8;
      uint4 pv = gv[j];
      const half_t* ph = (const half_t*)&pv;
#pragma unroll
      for (int t = 0; t < 8; ++t) sVT[wave][(w0 + t) * VLD + g] = ph[t];
    }
  }
  asm volatile("s_wait_asynccnt 0x0" ::: "memory");
  __syncthreads();

  // ---- S = Qs @ K^T : all fragments hoisted, 16 WMMAs per K-step ----
  const v8f vzero = {};
  v8f S[4][4];
#pragma unroll
  for (int it = 0; it < 4; ++it)
#pragma unroll
    for (int jt = 0; jt < 4; ++jt) S[it][jt] = vzero;

#pragma unroll
  for (int kt = 0; kt < 2; ++kt) {
    v16h aq[4], bfr[4];
#pragma unroll
    for (int it = 0; it < 4; ++it)
      aq[it] = load_a_frag(sQ[wave], WW, it * 16, kt * 32, lane);
#pragma unroll
    for (int jt = 0; jt < 4; ++jt)
      bfr[jt] = load_bt_frag(sK[wave], WW, kt * 32, jt * 16, lane);
#pragma unroll
    for (int it = 0; it < 4; ++it)
#pragma unroll
      for (int jt = 0; jt < 4; ++jt)
        S[it][jt] = wmma16(aq[it], bfr[jt], S[it][jt]);
  }

  // ---- softmax (scale already folded into q projection). C-layout rows live
  // across 16 lanes of a half-wave -> xor-shuffle reduce with masks 1..8 ----
#pragma unroll
  for (int it = 0; it < 4; ++it)
#pragma unroll
    for (int r = 0; r < 8; ++r) {
      float mx = -3.0e38f;
#pragma unroll
      for (int jt = 0; jt < 4; ++jt) mx = fmaxf(mx, S[it][jt][r]);
#pragma unroll
      for (int off = 1; off < 16; off <<= 1) mx = fmaxf(mx, __shfl_xor(mx, off));
      float sum = 0.f;
#pragma unroll
      for (int jt = 0; jt < 4; ++jt) {
        const float e = __expf(S[it][jt][r] - mx);
        S[it][jt][r] = e;
        sum += e;
      }
#pragma unroll
      for (int off = 1; off < 16; off <<= 1) sum += __shfl_xor(sum, off);
      const float inv = 1.0f / sum;
#pragma unroll
      for (int jt = 0; jt < 4; ++jt) S[it][jt][r] *= inv;
    }

  // ---- re-shape P through LDS (reuse Q buffer) into A-fragment layout ----
  __syncthreads();
#pragma unroll
  for (int it = 0; it < 4; ++it)
#pragma unroll
    for (int jt = 0; jt < 4; ++jt) {
      const int g  = jt * 16 + (lane & 15);
      const int hb = it * 16 + ((lane >> 4) << 3);
#pragma unroll
      for (int r = 0; r < 8; ++r)
        sQ[wave][(hb + r) * WW + g] = (half_t)S[it][jt][r];
    }
  __syncthreads();

  // ---- O = P @ V (B-fragments contiguous from transposed V) ----
  v8f O[4][4];
#pragma unroll
  for (int it = 0; it < 4; ++it)
#pragma unroll
    for (int nt = 0; nt < 4; ++nt) O[it][nt] = vzero;

#pragma unroll
  for (int kt = 0; kt < 2; ++kt) {
    v16h ap[4], bfr[4];
#pragma unroll
    for (int it = 0; it < 4; ++it)
      ap[it] = load_a_frag(sQ[wave], WW, it * 16, kt * 32, lane);
#pragma unroll
    for (int nt = 0; nt < 4; ++nt)
      bfr[nt] = load_bt_frag(sVT[wave], VLD, kt * 32, nt * 16, lane);
#pragma unroll
    for (int it = 0; it < 4; ++it)
#pragma unroll
      for (int nt = 0; nt < 4; ++nt)
        O[it][nt] = wmma16(ap[it], bfr[nt], O[it][nt]);
  }

  // ---- epilogue: out = O + x (f32 residual, coalesced across lanes) ----
#pragma unroll
  for (int it = 0; it < 4; ++it)
#pragma unroll
    for (int nt = 0; nt < 4; ++nt) {
      const int wcol = nt * 16 + (lane & 15);
      const int hb   = it * 16 + ((lane >> 4) << 3);
#pragma unroll
      for (int r = 0; r < 8; ++r) {
        const int idx = (hb + r) * WW + wcol;
        os[idx] = O[it][nt][r] + xs[idx];
      }
    }
}

// ---------------------------------------------------------------------------
extern "C" void kernel_launch(void* const* d_in, const int* in_sizes, int n_in,
                              void* d_out, int out_size, void* d_ws, size_t ws_size,
                              hipStream_t stream) {
  (void)in_sizes; (void)n_in; (void)out_size; (void)ws_size;
  const float* x   = (const float*)d_in[0];
  const float* ctx = (const float*)d_in[1];
  const float* wq  = (const float*)d_in[2];
  const float* bq  = (const float*)d_in[3];
  const float* wk  = (const float*)d_in[4];
  const float* bk  = (const float*)d_in[5];
  const float* wv  = (const float*)d_in[6];
  const float* bv  = (const float*)d_in[7];
  float* out = (float*)d_out;

  const size_t tensor_elems = (size_t)NB * DIMC * DHW;
  half_t* qws = (half_t*)d_ws;
  half_t* kws = qws + tensor_elems;
  half_t* vws = kws + tensor_elems;

  dim3 gp(DHW / 128, NB);
  proj16_kernel<<<gp, 256, 0, stream>>>(x,   wq, bq, qws, DIMC, 0.125f);
  proj16_kernel<<<gp, 256, 0, stream>>>(ctx, wk, bk, kws, CTXC, 1.0f);
  proj16_kernel<<<gp, 256, 0, stream>>>(ctx, wv, bv, vws, CTXC, 1.0f);

  dim3 ga(DIMC / 2, DD, NB);
  attn64_kernel<<<ga, 64, 0, stream>>>(qws, kws, vws, x, out);
}
